// Cifar10_GlobalFedCaps_51900384805407
// MI455X (gfx1250) — matrix operand order repaired and checked by reading, not compiled, not run on hardware
//
#include <hip/hip_runtime.h>
#include <hip/hip_bf16.h>
#include <cstdint>

typedef __attribute__((ext_vector_type(16))) __bf16 v16bf;
typedef __attribute__((ext_vector_type(8)))  __bf16 v8bf;
typedef __attribute__((ext_vector_type(8)))  float  v8f;
typedef __attribute__((ext_vector_type(2)))  float  v2f;
typedef __attribute__((ext_vector_type(4)))  int    v4i;

#define BB   128
#define EE   3
#define IIC  1536
#define NCL  10

// gfx1250 async global->LDS path (guarded; falls back to plain loads)
#if defined(__has_builtin)
#  if __has_builtin(__builtin_amdgcn_global_load_async_to_lds_b128) && \
      __has_builtin(__builtin_amdgcn_s_wait_asynccnt)
#    define USE_ASYNC_LDS 1
#  endif
#endif
#ifndef USE_ASYNC_LDS
#  define USE_ASYNC_LDS 0
#endif

#define AS1 __attribute__((address_space(1)))
#define AS3 __attribute__((address_space(3)))

// ---------------------------------------------------------------------------
// Weight repack: conv2_w / prim_w (E,CO,CI,9,9) f32 -> bf16 [e][kyx][co][ci]
// ---------------------------------------------------------------------------
__global__ __launch_bounds__(256) void k_convert_w(
    const float* __restrict__ c2w, const float* __restrict__ pw,
    __bf16* __restrict__ dst2, __bf16* __restrict__ dstp)
{
    int idx = blockIdx.x * 256 + threadIdx.x;          // < 3*81*256*256
    int ci  = idx & 255;
    int co  = (idx >> 8) & 255;
    int t   = idx >> 16;                               // e*81 + kyx
    int e   = t / 81;
    int kyx = t - e * 81;
    int src = ((e * 256 + co) * 256 + ci) * 81 + kyx;
    if (blockIdx.y == 0) dst2[idx] = (__bf16)c2w[src];
    else                 dstp[idx] = (__bf16)pw[src];
}

// ---------------------------------------------------------------------------
// conv1: 1->256, 9x9, VALID, relu. out NHWC bf16 (E,B,24,24,256)
// ---------------------------------------------------------------------------
__global__ __launch_bounds__(256) void k_conv1(
    const float* __restrict__ x, const float* __restrict__ w,
    const float* __restrict__ bias, __bf16* __restrict__ h1)
{
    __shared__ float img[1024];
    const int bi = blockIdx.x;
    const int e  = blockIdx.y;
    const int t  = threadIdx.x;                        // co
    ((float4*)img)[t] = ((const float4*)(x + (bi * 3 + e) * 1024))[t];
    float wreg[81];
    const float* wp = w + (e * 256 + t) * 81;
#pragma unroll
    for (int q = 0; q < 81; ++q) wreg[q] = wp[q];
    const float b0 = bias[e * 256 + t];
    __syncthreads();
    __bf16* outp = h1 + (size_t)((e * BB + bi) * 576) * 256 + t;
    for (int p = 0; p < 576; ++p) {
        int oy = p / 24, ox = p - oy * 24;
        float acc = b0;
#pragma unroll
        for (int ky = 0; ky < 9; ++ky)
#pragma unroll
            for (int kx = 0; kx < 9; ++kx)
                acc = fmaf(img[(oy + ky) * 32 + ox + kx], wreg[ky * 9 + kx], acc);
        outp[(size_t)p * 256] = (__bf16)fmaxf(acc, 0.f);
    }
}

// ---------------------------------------------------------------------------
// Implicit-GEMM 9x9 conv via bf16 WMMA.
//   D[n, co] = sum_{kyx,ci} act[b, oy*S+ky, ox*S+kx, ci] * W[e][kyx][co][ci]
// block tile 128(n) x 256(co); 8 waves, each 64x64; K-step = 32 ci.
// Double-buffered LDS, async global->LDS fill when available.
// ---------------------------------------------------------------------------
template<int IN_H, int OUT_H, int STRIDE, bool RELU_BF16>
__global__ __launch_bounds__(256) void k_conv_gemm(
    const __bf16* __restrict__ act, const __bf16* __restrict__ wt,
    const float* __restrict__ bias, __bf16* __restrict__ out_bf,
    float* __restrict__ out_caps)
{
    constexpr int LOG_OH = (OUT_H == 16) ? 4 : 2;
    constexpr int RS = 40;                         // padded LDS row (bf16), 80B
    __shared__ __bf16 At[2][128 * RS];             // 20 KB
    __shared__ __bf16 Bt[2][256 * RS];             // 40 KB

    const int e    = blockIdx.z;
    const int n0   = blockIdx.x * 128;
    const int tid  = threadIdx.x;
    const int lane = tid & 31;
    const int wave = tid >> 5;
    const int wn   = (wave & 1) * 64;              // n rows owned by wave
    const int wc   = (wave >> 1) * 64;             // co cols owned by wave

    // --- loader mapping ----------------------------------------------------
    // At: 2 threads per n-row, 16 ci (32B) each.  Bt: 1 thread per co-row, 32 ci.
    const int lra = tid >> 1;
    const int lca = (tid & 1) * 16;
    const int ln   = n0 + lra;
    const int lb   = ln >> (2 * LOG_OH);
    const int lrem = ln & (OUT_H * OUT_H - 1);
    const int loy  = lrem >> LOG_OH;
    const int lox  = lrem & (OUT_H - 1);
    const int act_row = (((e * BB + lb) * IN_H + loy * STRIDE) * IN_H + lox * STRIDE) * 256;
    const int wt_row  = e * 81 * 65536 + tid * 256;

    auto load_step = [&](int buf, int ks) {
        int kyx = ks >> 3;
        int ci0 = (ks & 7) << 5;
        int ky  = kyx / 9, kx = kyx - ky * 9;
        int aoff = act_row + (ky * IN_H + kx) * 256 + ci0 + lca;
        int woff = wt_row + kyx * 65536 + ci0;
#if USE_ASYNC_LDS
        __builtin_amdgcn_global_load_async_to_lds_b128(
            (AS1 v4i*)&act[aoff],      (AS3 v4i*)&At[buf][lra * RS + lca], 0, 0);
        __builtin_amdgcn_global_load_async_to_lds_b128(
            (AS1 v4i*)&act[aoff + 8],  (AS3 v4i*)&At[buf][lra * RS + lca + 8], 0, 0);
#pragma unroll
        for (int c = 0; c < 32; c += 8)
            __builtin_amdgcn_global_load_async_to_lds_b128(
                (AS1 v4i*)&wt[woff + c], (AS3 v4i*)&Bt[buf][tid * RS + c], 0, 0);
#else
        *(v8bf*)&At[buf][lra * RS + lca]     = *(const v8bf*)&act[aoff];
        *(v8bf*)&At[buf][lra * RS + lca + 8] = *(const v8bf*)&act[aoff + 8];
#pragma unroll
        for (int c = 0; c < 32; c += 8)
            *(v8bf*)&Bt[buf][tid * RS + c] = *(const v8bf*)&wt[woff + c];
#endif
    };

    // A fragment (16x32 bf16): lane<16 -> K 0..7 & 16..23, lane>=16 -> 8..15 & 24..31
    auto fragA = [&](const __bf16* T, int r0) -> v16bf {
        int row = r0 + (lane & 15);
        int k0  = (lane < 16) ? 0 : 8;
        v8bf lo = *(const v8bf*)&T[row * RS + k0];
        v8bf hi = *(const v8bf*)&T[row * RS + k0 + 16];
        return __builtin_shufflevector(lo, hi, 0,1,2,3,4,5,6,7,8,9,10,11,12,13,14,15);
    };
    // B fragment (32x16 bf16): lane<16 -> K 0..7 & 16..23, lane>=16 -> 8..15 & 24..31
    auto fragB = [&](const __bf16* T, int r0) -> v16bf {
        int row = r0 + (lane & 15);
        int k0  = (lane < 16) ? 0 : 8;
        v8bf lo = *(const v8bf*)&T[row * RS + k0];
        v8bf hi = *(const v8bf*)&T[row * RS + k0 + 16];
        return __builtin_shufflevector(lo, hi, 0,1,2,3,4,5,6,7,8,9,10,11,12,13,14,15);
    };

    v8f acc[4][4];
#pragma unroll
    for (int m = 0; m < 4; ++m)
#pragma unroll
        for (int ns = 0; ns < 4; ++ns) acc[m][ns] = (v8f)0.0f;

    constexpr int NK = 81 * 8;
    load_step(0, 0);
#if USE_ASYNC_LDS
    __builtin_amdgcn_s_wait_asynccnt(0);
#endif
    __syncthreads();
    for (int ks = 0; ks < NK; ++ks) {
        int buf = ks & 1;
        if (ks + 1 < NK) load_step(buf ^ 1, ks + 1);
        v16bf a0 = fragA(At[buf], wn);
        v16bf a1 = fragA(At[buf], wn + 16);
        v16bf a2 = fragA(At[buf], wn + 32);
        v16bf a3 = fragA(At[buf], wn + 48);
#pragma unroll
        for (int ns = 0; ns < 4; ++ns) {
            v16bf bfrag = fragB(Bt[buf], wc + ns * 16);
            acc[0][ns] = __builtin_amdgcn_wmma_f32_16x16x32_bf16(false, a0, false, bfrag, (short)0, acc[0][ns], false, false);
            acc[1][ns] = __builtin_amdgcn_wmma_f32_16x16x32_bf16(false, a1, false, bfrag, (short)0, acc[1][ns], false, false);
            acc[2][ns] = __builtin_amdgcn_wmma_f32_16x16x32_bf16(false, a2, false, bfrag, (short)0, acc[2][ns], false, false);
            acc[3][ns] = __builtin_amdgcn_wmma_f32_16x16x32_bf16(false, a3, false, bfrag, (short)0, acc[3][ns], false, false);
        }
#if USE_ASYNC_LDS
        __builtin_amdgcn_s_wait_asynccnt(0);
#endif
        __syncthreads();
    }

    // ---- epilogue: bias (+relu->bf16 NHWC) or scatter to caps (f32) -------
    const int rowhi = (lane >> 4) << 3;            // C/D layout: lanes>=16 -> M+8
    const int cn    = lane & 15;
#pragma unroll
    for (int m = 0; m < 4; ++m) {
#pragma unroll
        for (int ns = 0; ns < 4; ++ns) {
            int co = wc + ns * 16 + cn;
            float bi = bias[e * 256 + co];
#pragma unroll
            for (int v = 0; v < 8; ++v) {
                int ng  = n0 + wn + m * 16 + v + rowhi;
                float val = acc[m][ns][v] + bi;
                int b_  = ng >> (2 * LOG_OH);
                int rem = ng & (OUT_H * OUT_H - 1);
                int oy  = rem >> LOG_OH;
                int ox  = rem & (OUT_H - 1);
                if constexpr (RELU_BF16) {
                    val = fmaxf(val, 0.f);
                    out_bf[(size_t)(((e * BB + b_) * OUT_H + oy) * OUT_H + ox) * 256 + co] = (__bf16)val;
                } else {
                    int pc = co >> 3, d = co & 7;
                    int i  = e * 512 + pc * 16 + oy * 4 + ox;
                    out_caps[(size_t)(b_ * IIC + i) * 8 + d] = val;
                }
            }
        }
    }
}

// ---------------------------------------------------------------------------
// squash all (B*I) capsule vectors of length 8, in place
// ---------------------------------------------------------------------------
__global__ __launch_bounds__(256) void k_squash(float* __restrict__ caps)
{
    int idx = blockIdx.x * 256 + threadIdx.x;      // < 128*1536
    float* p = caps + (size_t)idx * 8;
    float v[8]; float l2 = 0.f;
#pragma unroll
    for (int d = 0; d < 8; ++d) { v[d] = p[d]; l2 = fmaf(v[d], v[d], l2); }
    float l  = sqrtf(fmaxf(l2, 1e-30f));
    float sc = l2 / ((1.f + l2) * l);
#pragma unroll
    for (int d = 0; d < 8; ++d) p[d] = v[d] * sc;
}

// ---------------------------------------------------------------------------
// u[b,i,j] = caps[b,i,:] @ caps_W[i,:,:]  via FP32 WMMA 16x16x4
// ---------------------------------------------------------------------------
__global__ __launch_bounds__(320) void k_caps_u(
    const float* __restrict__ caps, const float* __restrict__ capsW,
    float* __restrict__ u)
{
    const int i    = blockIdx.x;
    const int lane = threadIdx.x & 31;
    const int jt   = threadIdx.x >> 5;             // 0..9
    const int cn   = lane & 15;
    const int kb   = (lane < 16) ? 0 : 2;          // f32 A/B lane-half K split
    const int j    = jt * 16 + cn;

    v2f bf0, bf1;
    bf0.x = capsW[(size_t)(i * 8 + kb)     * 160 + j];
    bf0.y = capsW[(size_t)(i * 8 + kb + 1) * 160 + j];
    bf1.x = capsW[(size_t)(i * 8 + 4 + kb)     * 160 + j];
    bf1.y = capsW[(size_t)(i * 8 + 4 + kb + 1) * 160 + j];

    const int rowhi = (lane >> 4) << 3;
    for (int bt = 0; bt < 8; ++bt) {
        int bi = bt * 16 + cn;
        const float* cp = caps + (size_t)(bi * IIC + i) * 8;
        v2f a0, a1;
        a0.x = cp[kb];     a0.y = cp[kb + 1];
        a1.x = cp[4 + kb]; a1.y = cp[4 + kb + 1];
        v8f c = (v8f)0.0f;
        c = __builtin_amdgcn_wmma_f32_16x16x4_f32(false, a0, false, bf0, (short)0, c, false, false);
        c = __builtin_amdgcn_wmma_f32_16x16x4_f32(false, a1, false, bf1, (short)0, c, false, false);
#pragma unroll
        for (int v = 0; v < 8; ++v) {
            int bg = bt * 16 + v + rowhi;
            u[(size_t)(bg * IIC + i) * 160 + j] = c[v];
        }
    }
}

// ---------------------------------------------------------------------------
// dynamic routing: one block per batch element. bb logits in registers;
// softmax coeffs staged in 60KB LDS; u re-reads stay L2-resident.
// ---------------------------------------------------------------------------
__global__ __launch_bounds__(256) void k_routing(
    const float* __restrict__ u, const float* __restrict__ blog,
    float* __restrict__ out)
{
    __shared__ float cl[IIC * NCL];                // 61440 B
    __shared__ float sl[160];
    __shared__ float vl[160];
    __shared__ float scl[NCL];

    const int b   = blockIdx.x;
    const int tid = threadIdx.x;
    const float* ub = u + (size_t)b * IIC * 160;

    float bbr[6][NCL];
#pragma unroll
    for (int q = 0; q < 6; ++q) {
        int i = q * 256 + tid;
#pragma unroll
        for (int o = 0; o < NCL; ++o) bbr[q][o] = blog[i * NCL + o];
    }
    __syncthreads();

    for (int it = 0; it < 4; ++it) {
#pragma unroll
        for (int q = 0; q < 6; ++q) {
            int i = q * 256 + tid;
            const float* ur = ub + (size_t)i * 160;
            if (it > 0) {
#pragma unroll
                for (int o = 0; o < NCL; ++o) {
                    float dot = 0.f;
#pragma unroll
                    for (int d = 0; d < 16; ++d)
                        dot = fmaf(ur[o * 16 + d], vl[o * 16 + d], dot);
                    bbr[q][o] += dot;
                }
            }
            float mx = bbr[q][0];
#pragma unroll
            for (int o = 1; o < NCL; ++o) mx = fmaxf(mx, bbr[q][o]);
            float se = 0.f;
            float ex[NCL];
#pragma unroll
            for (int o = 0; o < NCL; ++o) { ex[o] = __expf(bbr[q][o] - mx); se += ex[o]; }
            float inv = 1.f / se;
#pragma unroll
            for (int o = 0; o < NCL; ++o) cl[i * NCL + o] = ex[o] * inv;
        }
        __syncthreads();
        if (tid < 160) {
            int o = tid >> 4;
            float s = 0.f;
            for (int i = 0; i < IIC; ++i)
                s = fmaf(cl[i * NCL + o], ub[(size_t)i * 160 + tid], s);
            sl[tid] = s;
        }
        __syncthreads();
        if (tid < NCL) {
            float l2 = 0.f;
#pragma unroll
            for (int d = 0; d < 16; ++d) l2 = fmaf(sl[tid * 16 + d], sl[tid * 16 + d], l2);
            float l = sqrtf(fmaxf(l2, 1e-30f));
            scl[tid] = l2 / ((1.f + l2) * l);
        }
        __syncthreads();
        if (tid < 160) vl[tid] = sl[tid] * scl[tid >> 4];
        __syncthreads();
    }

    if (tid < 160) out[(size_t)b * 160 + tid] = vl[tid];
    if (tid < NCL) {
        float l2 = 0.f;
#pragma unroll
        for (int d = 0; d < 16; ++d) l2 = fmaf(vl[tid * 16 + d], vl[tid * 16 + d], l2);
        out[(size_t)BB * 160 + b * NCL + tid] = sqrtf(l2);
    }
}

// ---------------------------------------------------------------------------
extern "C" void kernel_launch(void* const* d_in, const int* in_sizes, int n_in,
                              void* d_out, int out_size, void* d_ws, size_t ws_size,
                              hipStream_t stream)
{
    const float* x    = (const float*)d_in[0];
    const float* c1w  = (const float*)d_in[1];
    const float* c1b  = (const float*)d_in[2];
    const float* c2w  = (const float*)d_in[3];
    const float* c2b  = (const float*)d_in[4];
    const float* pw   = (const float*)d_in[5];
    const float* pb   = (const float*)d_in[6];
    const float* cW   = (const float*)d_in[7];
    const float* blog = (const float*)d_in[8];
    float* out = (float*)d_out;

    char* ws = (char*)d_ws;
    __bf16* Wc2 = (__bf16*)ws; ws += 31850496;     // [3][81][256][256] bf16
    __bf16* Wpr = (__bf16*)ws; ws += 31850496;
    __bf16* h1  = (__bf16*)ws; ws += 113246208;    // (3,128,24,24,256) bf16 NHWC
    __bf16* h2  = (__bf16*)ws; ws += 50331648;     // (3,128,16,16,256) bf16 NHWC
    float*  caps= (float*)ws;  ws += 6291456;      // (128,1536,8) f32
    float*  u   = (float*)ws;  ws += 125829120;    // (128,1536,160) f32

    hipLaunchKernelGGL(k_convert_w, dim3(62208, 2), dim3(256), 0, stream, c2w, pw, Wc2, Wpr);
    hipLaunchKernelGGL(k_conv1, dim3(128, 3), dim3(256), 0, stream, x, c1w, c1b, h1);
    hipLaunchKernelGGL((k_conv_gemm<24, 16, 1, true>), dim3(256, 1, 3), dim3(256), 0, stream,
                       h1, Wc2, c2b, h2, (float*)nullptr);
    hipLaunchKernelGGL((k_conv_gemm<16, 4, 2, false>), dim3(16, 1, 3), dim3(256), 0, stream,
                       h2, Wpr, pb, (__bf16*)nullptr, caps);
    hipLaunchKernelGGL(k_squash, dim3(768), dim3(256), 0, stream, caps);
    hipLaunchKernelGGL(k_caps_u, dim3(1536), dim3(320), 0, stream, caps, cW, u);
    hipLaunchKernelGGL(k_routing, dim3(128), dim3(256), 0, stream, u, blog, out);
}
